// AttentionBlock_66099546686059
// MI455X (gfx1250) — matrix operand order repaired and checked by reading, not compiled, not run on hardware
//
#include <hip/hip_runtime.h>

typedef __attribute__((ext_vector_type(16))) _Float16 v16h;
typedef __attribute__((ext_vector_type(8)))  float    v8f;

#define C_      512
#define N_      1024
#define B_      8
#define NH      8
#define HD      64
#define GROUPS_ 32
#define GSIZE   (C_ / GROUPS_)   // 16 channels per group
#define EPS_    1e-5f

union FragH { v16h h; uint4 q[2]; };

#define WMMA_F16(A, B, Cacc) \
  __builtin_amdgcn_wmma_f32_16x16x32_f16(false, (A), false, (B), (short)0, (Cacc), false, false)

// A fragment: 16(M) x 32(K) f16, source row-major (K contiguous), ld = row stride.
// ISA layout: lane m=lane%16; kh=lane>>4; K = kh*8+{0..7} then 16+kh*8+{0..7}.
__device__ inline FragH load_frag_a(const _Float16* base, int ld) {
  int lane = threadIdx.x & 31;
  const _Float16* p = base + (size_t)(lane & 15) * ld + (lane >> 4) * 8;
  FragH f;
  f.q[0] = *(const uint4*)(p);
  f.q[1] = *(const uint4*)(p + 16);
  return f;
}

// B fragment: 32(K) x 16(N) f16, source stored [n][k] (K contiguous), ld = n stride.
// ISA layout: lane n=lane%16; kh=lane>>4; K = kh*8+{0..7} then 16+kh*8+{0..7}.
__device__ inline FragH load_frag_b(const _Float16* base, int ld) {
  int lane = threadIdx.x & 31;
  const _Float16* p = base + (size_t)(lane & 15) * ld + (lane >> 4) * 8;
  FragH f;
  f.q[0] = *(const uint4*)(p);
  f.q[1] = *(const uint4*)(p + 16);
  return f;
}

// ---------------- K0: convert fp32 weights to f16 ----------------
__global__ void cvt_weights(const float* __restrict__ wq, const float* __restrict__ wp,
                            _Float16* __restrict__ wqh, _Float16* __restrict__ wph) {
  int i = blockIdx.x * 256 + threadIdx.x;
  if (i < 3 * C_ * C_) wqh[i] = (_Float16)wq[i];
  if (i < C_ * C_)     wph[i] = (_Float16)wp[i];
}

// ---------------- K1: GroupNorm -> xn transposed (B, N, C) f16 ----------------
__global__ void groupnorm_to_xnT(const float* __restrict__ x,
                                 const float* __restrict__ gw, const float* __restrict__ gb,
                                 _Float16* __restrict__ xnT) {
  int b = blockIdx.x >> 5;       // batch
  int g = blockIdx.x & 31;       // group
  const float* xg = x + ((size_t)b * C_ + g * GSIZE) * N_;   // contiguous 16384 floats
  int tid = threadIdx.x;

  float s = 0.f, ss = 0.f;
  for (int e = tid; e < GSIZE * N_; e += 256) {
    float v = xg[e]; s += v; ss += v * v;
  }
  __shared__ float rs[256], rq[256];
  rs[tid] = s; rq[tid] = ss;
  __syncthreads();
  for (int off = 128; off > 0; off >>= 1) {
    if (tid < off) { rs[tid] += rs[tid + off]; rq[tid] += rq[tid + off]; }
    __syncthreads();
  }
  const float inv = 1.f / (GSIZE * N_);
  float mean = rs[0] * inv;
  float var  = rq[0] * inv - mean * mean;
  float rstd = rsqrtf(var + EPS_);

  float wv[GSIZE], bv[GSIZE];
  for (int cl = 0; cl < GSIZE; ++cl) {
    float wc = gw[g * GSIZE + cl];
    wv[cl] = wc * rstd;
    bv[cl] = gb[g * GSIZE + cl] - mean * rstd * wc;
  }
  _Float16* dst = xnT + (size_t)b * N_ * C_ + g * GSIZE;
  for (int n = tid; n < N_; n += 256) {
    union { _Float16 h[16]; uint4 q[2]; } o;
    for (int cl = 0; cl < GSIZE; ++cl)
      o.h[cl] = (_Float16)(xg[cl * N_ + n] * wv[cl] + bv[cl]);
    uint4* dp = (uint4*)(dst + (size_t)n * C_);
    dp[0] = o.q[0]; dp[1] = o.q[1];
  }
}

// ---------- epilogue helper: route one 16x16 D tile to qT/kT (scaled) or v ----------
__device__ inline void store_qkv_tile(const v8f& acc, int otile, int ncol_base, int b,
                                      _Float16* __restrict__ qT, _Float16* __restrict__ kT,
                                      _Float16* __restrict__ vv) {
  int lane = threadIdx.x & 31;
  int hi   = lane >> 4;
  int ncol = ncol_base + (lane & 15);
  int hh = otile / (3 * HD);
  int r  = otile % (3 * HD);
  int bh = b * NH + hh;
  const float scale = 0.35355339059327379f;   // 64^-0.25, applied to q and k
  if (r < 2 * HD) {
    _Float16* dst = (r < HD) ? qT : kT;
    int c0 = (r & (HD - 1)) + hi * 8;
    union { _Float16 h[8]; uint4 q; } pk;
    for (int i = 0; i < 8; ++i) pk.h[i] = (_Float16)(acc[i] * scale);
    *(uint4*)(dst + ((size_t)bh * N_ + ncol) * HD + c0) = pk.q;
  } else {
    int c0 = (r - 2 * HD) + hi * 8;
    for (int i = 0; i < 8; ++i)
      vv[((size_t)bh * HD + c0 + i) * N_ + ncol] = (_Float16)acc[i];
  }
}

// ------- K2: QKV GEMM, 2x2 register-blocked WMMA (4 WMMA per 8 b128 loads) -------
__global__ void __launch_bounds__(128)
qkv_gemm(const _Float16* __restrict__ wqh, const _Float16* __restrict__ xnT,
         _Float16* __restrict__ qT, _Float16* __restrict__ kT,
         _Float16* __restrict__ vv) {
  int b  = blockIdx.z;
  int og = blockIdx.y;            // 12 groups of 128 output channels
  int nt = blockIdx.x;            // 32 tiles of 32 columns
  int w  = threadIdx.x >> 5;      // wave -> 32-row sub-block
  int o0 = og * 128 + w * 32;
  int n0 = nt * 32;

  v8f acc00 = {}, acc01 = {}, acc10 = {}, acc11 = {};
  const _Float16* a0 = wqh + (size_t)o0 * C_;
  const _Float16* a1 = a0 + (size_t)16 * C_;
  const _Float16* b0 = xnT + ((size_t)b * N_ + n0) * C_;
  const _Float16* b1 = b0 + (size_t)16 * C_;
  for (int kk = 0; kk < C_; kk += 32) {
    FragH A0 = load_frag_a(a0 + kk, C_);
    FragH A1 = load_frag_a(a1 + kk, C_);
    FragH B0 = load_frag_b(b0 + kk, C_);
    FragH B1 = load_frag_b(b1 + kk, C_);
    acc00 = WMMA_F16(A0.h, B0.h, acc00);
    acc01 = WMMA_F16(A0.h, B1.h, acc01);
    acc10 = WMMA_F16(A1.h, B0.h, acc10);
    acc11 = WMMA_F16(A1.h, B1.h, acc11);
  }
  store_qkv_tile(acc00, o0,      n0,      b, qT, kT, vv);
  store_qkv_tile(acc01, o0,      n0 + 16, b, qT, kT, vv);
  store_qkv_tile(acc10, o0 + 16, n0,      b, qT, kT, vv);
  store_qkv_tile(acc11, o0 + 16, n0 + 16, b, qT, kT, vv);
}

// ---------------- K3: attention: S=qT*k (WMMA) -> softmax -> h=v*P^T (WMMA) ----------
__global__ void __launch_bounds__(128)
attention_kernel(const _Float16* __restrict__ qT, const _Float16* __restrict__ kT,
                 const _Float16* __restrict__ vv, _Float16* __restrict__ hT) {
  __shared__ __attribute__((aligned(16))) float    S[16 * N_];   // 64 KB logits
  __shared__ __attribute__((aligned(16))) _Float16 P[16 * N_];   // 32 KB exp(S-max)
  __shared__ float rsum[16];

  int bh = blockIdx.x;            // 64 head-batches
  int t0 = blockIdx.y * 16;       // 64 t-tiles
  int w    = threadIdx.x >> 5;
  int lane = threadIdx.x & 31;
  int m  = lane & 15;
  int kh = lane >> 4;

  // q A-fragments for K-steps c=0..31, 32..63 (kept in registers)
  FragH qa0, qa1;
  {
    const _Float16* p0 = qT + ((size_t)bh * N_ + t0 + m) * HD + kh * 8;
    qa0.q[0] = *(const uint4*)(p0);      qa0.q[1] = *(const uint4*)(p0 + 16);
    qa1.q[0] = *(const uint4*)(p0 + 32); qa1.q[1] = *(const uint4*)(p0 + 48);
  }
  // Phase 1: S[t, s] for all 1024 s (each wave covers 16 of the 64 s-tiles)
  const _Float16* kbase = kT + (size_t)bh * N_ * HD;
  for (int st = w; st < 64; st += 4) {
    int s0 = st * 16;
    const _Float16* pb = kbase + (size_t)(s0 + m) * HD + kh * 8;
    FragH b0, b1;
    b0.q[0] = *(const uint4*)(pb);      b0.q[1] = *(const uint4*)(pb + 16);
    b1.q[0] = *(const uint4*)(pb + 32); b1.q[1] = *(const uint4*)(pb + 48);
    v8f acc = {};
    acc = WMMA_F16(qa0.h, b0.h, acc);
    acc = WMMA_F16(qa1.h, b1.h, acc);
    int scol = s0 + m;
    for (int i = 0; i < 8; ++i)
      S[(i + 8 * kh) * N_ + scol] = acc[i];
  }
  __syncthreads();

  // Phase 2: row softmax (8 lanes per row, butterfly shuffles; P left unnormalized)
  {
    int r = threadIdx.x >> 3;
    int j = threadIdx.x & 7;
    float mx = -1e30f;
    for (int s = j; s < N_; s += 8) mx = fmaxf(mx, S[r * N_ + s]);
    for (int d = 1; d < 8; d <<= 1) mx = fmaxf(mx, __shfl_xor(mx, d, 8));
    float sum = 0.f;
    for (int s = j; s < N_; s += 8) {
      float p = __expf(S[r * N_ + s] - mx);
      sum += p;
      P[r * N_ + s] = (_Float16)p;
    }
    for (int d = 1; d < 8; d <<= 1) sum += __shfl_xor(sum, d, 8);
    if (j == 0) rsum[r] = 1.f / sum;
  }
  __syncthreads();

  // Phase 3: h[c, t] = sum_s v[c,s] * P[t,s]; wave w handles c-tile w
  v8f acc = {};
  const _Float16* vrow = vv + ((size_t)bh * HD + w * 16 + m) * N_ + kh * 8;
  const _Float16* prow = &P[m * N_ + kh * 8];
  for (int st = 0; st < 32; ++st) {
    FragH a, bf;
    const _Float16* pa = vrow + st * 32;
    a.q[0]  = *(const uint4*)(pa);  a.q[1]  = *(const uint4*)(pa + 16);
    const _Float16* pb = prow + st * 32;
    bf.q[0] = *(const uint4*)(pb);  bf.q[1] = *(const uint4*)(pb + 16);
    acc = WMMA_F16(a.h, bf.h, acc);
  }
  float rsc = rsum[m];                      // 1/rowsum for this t column
  int b  = bh >> 3, hh = bh & 7;
  int tcol = t0 + m;
  int c0 = hh * HD + w * 16 + kh * 8;
  union { _Float16 h[8]; uint4 q; } pk;
  for (int i = 0; i < 8; ++i) pk.h[i] = (_Float16)(acc[i] * rsc);
  *(uint4*)(hT + ((size_t)b * N_ + tcol) * C_ + c0) = pk.q;
}

// ------- K4: proj GEMM, 2x2 register-blocked WMMA, + bias + residual -> f32 out -------
__global__ void __launch_bounds__(128)
proj_gemm(const _Float16* __restrict__ wph, const _Float16* __restrict__ hT,
          const float* __restrict__ pb, const float* __restrict__ x,
          float* __restrict__ out) {
  int b  = blockIdx.z;
  int og = blockIdx.y;            // 4 groups of 128 output channels
  int nt = blockIdx.x;            // 32 tiles of 32 columns
  int w    = threadIdx.x >> 5;
  int lane = threadIdx.x & 31;
  int o0 = og * 128 + w * 32;
  int n0 = nt * 32;

  v8f acc00 = {}, acc01 = {}, acc10 = {}, acc11 = {};
  const _Float16* a0 = wph + (size_t)o0 * C_;
  const _Float16* a1 = a0 + (size_t)16 * C_;
  const _Float16* b0 = hT + ((size_t)b * N_ + n0) * C_;
  const _Float16* b1 = b0 + (size_t)16 * C_;
  for (int kk = 0; kk < C_; kk += 32) {
    FragH A0 = load_frag_a(a0 + kk, C_);
    FragH A1 = load_frag_a(a1 + kk, C_);
    FragH B0 = load_frag_b(b0 + kk, C_);
    FragH B1 = load_frag_b(b1 + kk, C_);
    acc00 = WMMA_F16(A0.h, B0.h, acc00);
    acc01 = WMMA_F16(A0.h, B1.h, acc01);
    acc10 = WMMA_F16(A1.h, B0.h, acc10);
    acc11 = WMMA_F16(A1.h, B1.h, acc11);
  }
  int hi = lane >> 4;
  const v8f* accs[4] = { &acc00, &acc01, &acc10, &acc11 };
  for (int t = 0; t < 4; ++t) {
    int otile = o0 + (t >> 1) * 16;
    int ncol  = n0 + (t & 1) * 16 + (lane & 15);
    const v8f& a = *accs[t];
    for (int i = 0; i < 8; ++i) {
      int o = otile + hi * 8 + i;
      size_t idx = ((size_t)b * C_ + o) * N_ + ncol;
      out[idx] = a[i] + pb[o] + x[idx];
    }
  }
}

extern "C" void kernel_launch(void* const* d_in, const int* in_sizes, int n_in,
                              void* d_out, int out_size, void* d_ws, size_t ws_size,
                              hipStream_t stream) {
  const float* x      = (const float*)d_in[0];
  const float* gn_w   = (const float*)d_in[1];
  const float* gn_b   = (const float*)d_in[2];
  const float* qkv_w  = (const float*)d_in[3];
  const float* proj_w = (const float*)d_in[4];
  const float* proj_b = (const float*)d_in[5];
  // d_in[6] = num_heads (fixed at 8)

  char* ws = (char*)d_ws;
  _Float16* xnT = (_Float16*)(ws);                        // 8 MB (B,N,C); reused as hT
  _Float16* qT  = (_Float16*)(ws + (8u  << 20));          // 8 MB (BH,N,HD), scaled
  _Float16* kT  = (_Float16*)(ws + (16u << 20));          // 8 MB (BH,N,HD), scaled
  _Float16* vv  = (_Float16*)(ws + (24u << 20));          // 8 MB (BH,HD,N)
  _Float16* wqh = (_Float16*)(ws + (32u << 20));          // 1.5 MB
  _Float16* wph = (_Float16*)(ws + (32u << 20) + (size_t)3 * C_ * C_ * 2); // 0.5 MB
  _Float16* hT  = xnT;                                    // xn fully consumed before K3
  float* out = (float*)d_out;

  cvt_weights<<<(3 * C_ * C_ + 255) / 256, 256, 0, stream>>>(qkv_w, proj_w, wqh, wph);
  groupnorm_to_xnT<<<B_ * GROUPS_, 256, 0, stream>>>(x, gn_w, gn_b, xnT);
  qkv_gemm<<<dim3(32, 12, B_), 128, 0, stream>>>(wqh, xnT, qT, kT, vv);
  attention_kernel<<<dim3(64, 64), 128, 0, stream>>>(qT, kT, vv, hT);
  proj_gemm<<<dim3(32, 4, B_), 128, 0, stream>>>(wph, hT, proj_b, x, out);
}
